// CrissCrossAttention_11931419148310
// MI455X (gfx1250) — matrix operand order repaired and checked by reading, not compiled, not run on hardware
//
#include <hip/hip_runtime.h>

// ---------------------------------------------------------------------------
// CrissCrossAttention for MI455X (gfx1250): wave32 + v_wmma_f32_16x16x32_f16
// ---------------------------------------------------------------------------

typedef __attribute__((ext_vector_type(16))) _Float16 v16h;
typedef __attribute__((ext_vector_type(8)))  _Float16 v8h;
typedef __attribute__((ext_vector_type(2)))  _Float16 v2h;
typedef __attribute__((ext_vector_type(8)))  float    v8f;
typedef __attribute__((ext_vector_type(4)))  float    v4f;

#define WMMA_F16(a, b, c) \
  __builtin_amdgcn_wmma_f32_16x16x32_f16(false, (a), false, (b), (short)0, (c), false, false)

#define Bn   16
#define Cc   512
#define Hh   96
#define Ww   96
#define CQK  64
#define HW   (Hh * Ww)   // 9216

// --- WMMA fragment loaders (wave32 layouts per CDNA5 ISA 7.12.2) ------------
// A fragment: 16(M) x 32(K); memory: M-major rows, contiguous K, stride rs (halves).
__device__ __forceinline__ v16h load_frag_a(const _Float16* base, int rs) {
  int l = threadIdx.x & 31;
  const _Float16* p = base + (l & 15) * rs + ((l >> 4) << 3);
  v8h lo = *(const v8h*)(p);
  v8h hi = *(const v8h*)(p + 16);
  v16h r;
#pragma unroll
  for (int i = 0; i < 8; ++i) { r[i] = lo[i]; r[i + 8] = hi[i]; }
  return r;
}

// B fragment: 32(K) x 16(N); memory as B^T: N-major rows, contiguous K, stride rs.
__device__ __forceinline__ v16h load_frag_b(const _Float16* base, int rs) {
  int l = threadIdx.x & 31;
  const _Float16* p = base + (l & 15) * rs + ((l >> 4) << 3);
  v8h lo = *(const v8h*)(p);
  v8h hi = *(const v8h*)(p + 16);
  v16h r;
#pragma unroll
  for (int i = 0; i < 8; ++i) { r[i] = lo[i]; r[i + 8] = hi[i]; }
  return r;
}

// --- shared tile-staging helpers for the projection GEMMs -------------------
// A tile: 64(M) x 32(K) f32 -> f16, one v8h (16B) LDS store per thread
__device__ __forceinline__ void stage_A(_Float16 (*As)[40], const float* __restrict__ Wm,
                                        int m0, int k0, int t) {
  int i = t >> 2, j0 = (t & 3) << 3;
  const float* src = Wm + (size_t)(m0 + i) * Cc + k0 + j0;
  v4f s0 = *(const v4f*)(src);
  v4f s1 = *(const v4f*)(src + 4);
  v8h st;
#pragma unroll
  for (int r = 0; r < 4; ++r) { st[r] = (_Float16)s0[r]; st[r + 4] = (_Float16)s1[r]; }
  *(v8h*)&As[i][j0] = st;
}

// B tile: x[b][k0..k0+31][p0..p0+127] -> Bs[n][k] transposed; each thread
// handles 2 adjacent K rows x 8 N, so stores are packed b32 (K pairs).
__device__ __forceinline__ void stage_B(_Float16 (*Bs)[40], const float* __restrict__ X,
                                        int b, int k0, int p0, int t) {
  int kk  = (t >> 4) << 1;        // even K row: 0,2,...,30
  int nn0 = (t & 15) << 3;        // 0..120
  const float* s0 = X + ((size_t)b * Cc + k0 + kk) * HW + p0 + nn0;
  const float* s1 = s0 + HW;      // next K row
  v4f a0 = *(const v4f*)(s0), a1 = *(const v4f*)(s0 + 4);
  v4f b0 = *(const v4f*)(s1), b1 = *(const v4f*)(s1 + 4);
#pragma unroll
  for (int n = 0; n < 4; ++n)
    *(v2h*)&Bs[nn0 + n][kk] = (v2h){(_Float16)a0[n], (_Float16)b0[n]};
#pragma unroll
  for (int n = 0; n < 4; ++n)
    *(v2h*)&Bs[nn0 + 4 + n][kk] = (v2h){(_Float16)a1[n], (_Float16)b1[n]};
}

// ---------------------------------------------------------------------------
// K1a: q and k projections in ONE launch (shares the streamed x slab in L2).
// grid (HW/128, 2, B): y==0 -> q, y==1 -> k.  M=64, out channels-last f16.
// ---------------------------------------------------------------------------
__global__ __launch_bounds__(256)
void k_proj_qk(const float* __restrict__ X,
               const float* __restrict__ Wq, const float* __restrict__ bq,
               _Float16* __restrict__ qt,
               const float* __restrict__ Wk, const float* __restrict__ bk,
               _Float16* __restrict__ kt) {
  __shared__ _Float16 As[64][40];
  __shared__ _Float16 Bs[128][40];
  const int b  = blockIdx.z;
  const int p0 = blockIdx.x * 128;
  const float* Wm   = (blockIdx.y == 0) ? Wq : Wk;
  const float* bias = (blockIdx.y == 0) ? bq : bk;
  _Float16*    outp = (blockIdx.y == 0) ? qt : kt;
  const int t  = threadIdx.x;
  const int wv = t >> 5;
  const int mi = wv & 3;
  const int njb = (wv >> 2) * 4;

  v8f acc[4] = {};
  for (int k0 = 0; k0 < Cc; k0 += 32) {
    stage_A(As, Wm, 0, k0, t);
    stage_B(Bs, X, b, k0, p0, t);
    __syncthreads();
    v16h a = load_frag_a(&As[mi * 16][0], 40);
#pragma unroll
    for (int j = 0; j < 4; ++j) {
      v16h bb = load_frag_b(&Bs[(njb + j) * 16][0], 40);
      acc[j] = WMMA_F16(a, bb, acc[j]);
    }
    __syncthreads();
  }
  const int l  = t & 31;
  const int o0 = mi * 16 + ((l >> 4) << 3);
  float bv[8];
#pragma unroll
  for (int r = 0; r < 8; ++r) bv[r] = bias[o0 + r];
#pragma unroll
  for (int j = 0; j < 4; ++j) {
    int p = p0 + (njb + j) * 16 + (l & 15);
    v8h st;
#pragma unroll
    for (int r = 0; r < 8; ++r) st[r] = (_Float16)(acc[j][r] + bv[r]);
    *(v8h*)(outp + ((size_t)b * HW + p) * CQK + o0) = st;
  }
}

// ---------------------------------------------------------------------------
// K1b: v projection: writes v_std (B,C,H,W) and vt (B,W,C,H), both f16.
// grid (HW/128, 8, B)
// ---------------------------------------------------------------------------
__global__ __launch_bounds__(256)
void k_proj_v(const float* __restrict__ X, const float* __restrict__ Wv,
              const float* __restrict__ bias, _Float16* __restrict__ vstd,
              _Float16* __restrict__ vt) {
  __shared__ _Float16 As[64][40];
  __shared__ _Float16 Bs[128][40];
  const int b  = blockIdx.z;
  const int m0 = blockIdx.y * 64;
  const int p0 = blockIdx.x * 128;
  const int t  = threadIdx.x;
  const int wv = t >> 5;
  const int mi = wv & 3;
  const int njb = (wv >> 2) * 4;

  v8f acc[4] = {};
  for (int k0 = 0; k0 < Cc; k0 += 32) {
    stage_A(As, Wv, m0, k0, t);
    stage_B(Bs, X, b, k0, p0, t);
    __syncthreads();
    v16h a = load_frag_a(&As[mi * 16][0], 40);
#pragma unroll
    for (int j = 0; j < 4; ++j) {
      v16h bb = load_frag_b(&Bs[(njb + j) * 16][0], 40);
      acc[j] = WMMA_F16(a, bb, acc[j]);
    }
    __syncthreads();
  }
  const int l  = t & 31;
  const int o0 = m0 + mi * 16 + ((l >> 4) << 3);
  float bv[8];
#pragma unroll
  for (int r = 0; r < 8; ++r) bv[r] = bias[o0 + r];
#pragma unroll
  for (int j = 0; j < 4; ++j) {
    int p = p0 + (njb + j) * 16 + (l & 15);
    int h = p / Ww, w = p % Ww;
#pragma unroll
    for (int r = 0; r < 8; ++r) {
      _Float16 val = (_Float16)(acc[j][r] + bv[r]);
      vstd[((size_t)b * Cc + o0 + r) * HW + p] = val;            // (B,C,H,W)
      vt[(((size_t)b * Ww + w) * Cc + o0 + r) * Hh + h] = val;   // (B,W,C,H)
    }
  }
}

// ---------------------------------------------------------------------------
// K2: energies. z=0: eH per (b,w): e[u,h] = k(u,w).q(h,w)   (K=64)
//               z=1: eW per (b,h): e[v,w'] = k(h,v).q(h,w')
// Output f32, N-major / M-contiguous: eHt(B,W,h,u) / eWt(B,H,w,v).
// grid (96, B, 2), block 128 (4 waves, each a 3x3 block of 16x16 tiles)
// ---------------------------------------------------------------------------
__global__ __launch_bounds__(128)
void k_energy(const _Float16* __restrict__ kt, const _Float16* __restrict__ qt,
              float* __restrict__ eHt, float* __restrict__ eWt) {
  const int s = blockIdx.x, b = blockIdx.y, mode = blockIdx.z;
  const _Float16 *Ab, *Bb;
  int As_, Bs_;
  float* Out;
  if (mode == 0) {
    Ab  = kt + ((size_t)b * HW + s) * CQK;  As_ = Ww * CQK;
    Bb  = qt + ((size_t)b * HW + s) * CQK;  Bs_ = Ww * CQK;
    Out = eHt + ((size_t)b * Ww + s) * (size_t)(Hh * 96);
  } else {
    Ab  = kt + ((size_t)b * Hh + s) * (size_t)(Ww * CQK);  As_ = CQK;
    Bb  = qt + ((size_t)b * Hh + s) * (size_t)(Ww * CQK);  Bs_ = CQK;
    Out = eWt + ((size_t)b * Hh + s) * (size_t)(Ww * 96);
  }
  const int wv  = threadIdx.x >> 5;
  const int mib = (wv & 1) * 3, njb = (wv >> 1) * 3;

  v8f acc[3][3] = {};
#pragma unroll
  for (int ks = 0; ks < 2; ++ks) {
    v16h a[3], bb[3];
#pragma unroll
    for (int i = 0; i < 3; ++i) a[i]  = load_frag_a(Ab + (size_t)(mib + i) * 16 * As_ + ks * 32, As_);
#pragma unroll
    for (int j = 0; j < 3; ++j) bb[j] = load_frag_b(Bb + (size_t)(njb + j) * 16 * Bs_ + ks * 32, Bs_);
#pragma unroll
    for (int i = 0; i < 3; ++i)
#pragma unroll
      for (int j = 0; j < 3; ++j) acc[i][j] = WMMA_F16(a[i], bb[j], acc[i][j]);
  }
  const int l = threadIdx.x & 31;
#pragma unroll
  for (int i = 0; i < 3; ++i)
#pragma unroll
    for (int j = 0; j < 3; ++j) {
      int m = (mib + i) * 16 + ((l >> 4) << 3);
      int n = (njb + j) * 16 + (l & 15);
      float* o = Out + (size_t)n * 96 + m;
      v8f A = acc[i][j];
      *(v4f*)(o)     = (v4f){A[0], A[1], A[2], A[3]};
      *(v4f*)(o + 4) = (v4f){A[4], A[5], A[6], A[7]};
    }
}

// ---------------------------------------------------------------------------
// K3: per-pixel softmax over 192 logits (eH with u==h masked, + eW), f16 out.
// float4 streaming loads, two passes, packed v8h stores.
// ---------------------------------------------------------------------------
__global__ __launch_bounds__(256)
void k_softmax(const float* __restrict__ eHt, const float* __restrict__ eWt,
               _Float16* __restrict__ aHt, _Float16* __restrict__ aWt) {
  int idx = blockIdx.x * 256 + threadIdx.x;          // = b*HW + h*W + w
  int b = idx / HW, hw = idx % HW, h = hw / Ww, w = hw % Ww;
  const float* eh = eHt + (((size_t)b * Ww + w) * Hh + h) * 96;
  const float* ew = eWt + (((size_t)b * Hh + h) * Ww + w) * 96;

  float mx = -3.4e38f;
#pragma unroll
  for (int u4 = 0; u4 < 24; ++u4) {
    v4f e = *(const v4f*)(eh + u4 * 4);
#pragma unroll
    for (int j = 0; j < 4; ++j) if (u4 * 4 + j != h) mx = fmaxf(mx, e[j]);
  }
#pragma unroll
  for (int v4i = 0; v4i < 24; ++v4i) {
    v4f e = *(const v4f*)(ew + v4i * 4);
#pragma unroll
    for (int j = 0; j < 4; ++j) mx = fmaxf(mx, e[j]);
  }
  float sum = 0.f;
#pragma unroll
  for (int u4 = 0; u4 < 24; ++u4) {
    v4f e = *(const v4f*)(eh + u4 * 4);
#pragma unroll
    for (int j = 0; j < 4; ++j) if (u4 * 4 + j != h) sum += __expf(e[j] - mx);
  }
#pragma unroll
  for (int v4i = 0; v4i < 24; ++v4i) {
    v4f e = *(const v4f*)(ew + v4i * 4);
#pragma unroll
    for (int j = 0; j < 4; ++j) sum += __expf(e[j] - mx);
  }
  float inv = 1.f / sum;

  _Float16* oh = aHt + (((size_t)b * Ww + w) * Hh + h) * 96;
  _Float16* ow = aWt + (((size_t)b * Hh + h) * Ww + w) * 96;
#pragma unroll
  for (int u8 = 0; u8 < 12; ++u8) {
    v4f e0 = *(const v4f*)(eh + u8 * 8);
    v4f e1 = *(const v4f*)(eh + u8 * 8 + 4);
    v8h st;
#pragma unroll
    for (int j = 0; j < 4; ++j) {
      st[j]     = (u8 * 8 + j == h)     ? (_Float16)0.f : (_Float16)(__expf(e0[j] - mx) * inv);
      st[j + 4] = (u8 * 8 + 4 + j == h) ? (_Float16)0.f : (_Float16)(__expf(e1[j] - mx) * inv);
    }
    *(v8h*)(oh + u8 * 8) = st;
  }
#pragma unroll
  for (int v8i = 0; v8i < 12; ++v8i) {
    v4f e0 = *(const v4f*)(ew + v8i * 8);
    v4f e1 = *(const v4f*)(ew + v8i * 8 + 4);
    v8h st;
#pragma unroll
    for (int j = 0; j < 4; ++j) {
      st[j]     = (_Float16)(__expf(e0[j] - mx) * inv);
      st[j + 4] = (_Float16)(__expf(e1[j] - mx) * inv);
    }
    *(v8h*)(ow + v8i * 8) = st;
  }
}

// ---------------------------------------------------------------------------
// K4: column aggregation per (b,w): tmpH[c,h] = sum_u vt[c,u] * aH[u,h]
// grid (96, B), block 256 (8 waves x 4 M-blocks x 6 N-blocks, K=96)
// ---------------------------------------------------------------------------
__global__ __launch_bounds__(256)
void k_aggH(const _Float16* __restrict__ vt, const _Float16* __restrict__ aHt,
            float* __restrict__ tmpH) {
  const int w = blockIdx.x, b = blockIdx.y;
  const _Float16* A0 = vt  + ((size_t)b * Ww + w) * (size_t)(Cc * Hh);
  const _Float16* B0 = aHt + ((size_t)b * Ww + w) * (size_t)(Hh * 96);
  float*          O0 = tmpH + ((size_t)b * Ww + w) * (size_t)(Cc * Hh);
  const int wv = threadIdx.x >> 5, l = threadIdx.x & 31;

  for (int mi = wv * 4; mi < wv * 4 + 4; ++mi) {
    v8f acc[6] = {};
#pragma unroll
    for (int ks = 0; ks < 3; ++ks) {
      v16h a = load_frag_a(A0 + (size_t)(mi * 16) * Hh + ks * 32, Hh);
#pragma unroll
      for (int j = 0; j < 6; ++j) {
        v16h bb = load_frag_b(B0 + (size_t)(j * 16) * 96 + ks * 32, 96);
        acc[j] = WMMA_F16(a, bb, acc[j]);
      }
    }
    const int c0 = mi * 16 + ((l >> 4) << 3);
#pragma unroll
    for (int j = 0; j < 6; ++j) {
      int hN = j * 16 + (l & 15);
#pragma unroll
      for (int r = 0; r < 8; ++r) O0[(size_t)(c0 + r) * Hh + hN] = acc[j][r];
    }
  }
}

// ---------------------------------------------------------------------------
// K5: row aggregation per (b,h): out = x + gamma * sum_v v[c,v]*aW[v,w']
// ---------------------------------------------------------------------------
__global__ __launch_bounds__(256)
void k_aggW(const _Float16* __restrict__ vstd, const _Float16* __restrict__ aWt,
            const float* __restrict__ X, const float* __restrict__ gamma,
            float* __restrict__ out) {
  const int h = blockIdx.x, b = blockIdx.y;
  const _Float16* A0 = vstd + (size_t)b * Cc * HW + (size_t)h * Ww;
  const _Float16* B0 = aWt + ((size_t)b * Hh + h) * (size_t)(Ww * 96);
  const float g = gamma[0];
  const int wv = threadIdx.x >> 5, l = threadIdx.x & 31;

  for (int mi = wv * 4; mi < wv * 4 + 4; ++mi) {
    v8f acc[6] = {};
#pragma unroll
    for (int ks = 0; ks < 3; ++ks) {
      v16h a = load_frag_a(A0 + (size_t)(mi * 16) * HW + ks * 32, HW);
#pragma unroll
      for (int j = 0; j < 6; ++j) {
        v16h bb = load_frag_b(B0 + (size_t)(j * 16) * 96 + ks * 32, 96);
        acc[j] = WMMA_F16(a, bb, acc[j]);
      }
    }
    const int c0 = mi * 16 + ((l >> 4) << 3);
#pragma unroll
    for (int j = 0; j < 6; ++j) {
      int wN = j * 16 + (l & 15);
#pragma unroll
      for (int r = 0; r < 8; ++r) {
        size_t oidx = (((size_t)b * Cc + c0 + r) * Hh + h) * Ww + wN;
        out[oidx] = X[oidx] + g * acc[j][r];
      }
    }
  }
}

// ---------------------------------------------------------------------------
// K6: out(B,C,H,W) += gamma * tmpH(B,W,C,H)  -- tiled LDS transpose
// grid (3, 3, B*C), block 256 (as 32x8)
// ---------------------------------------------------------------------------
__global__ __launch_bounds__(256)
void k_transpose_add(const float* __restrict__ tmpH, const float* __restrict__ gamma,
                     float* __restrict__ out) {
  __shared__ float tile[32][33];
  const int bc = blockIdx.z;
  const int b = bc / Cc, c = bc % Cc;
  const int w0 = blockIdx.x * 32, h0 = blockIdx.y * 32;
  const int tx = threadIdx.x & 31, ty = threadIdx.x >> 5;
  const float* src = tmpH + (size_t)b * Ww * Cc * Hh;
  for (int i = ty; i < 32; i += 8)
    tile[i][tx] = src[((size_t)(w0 + i) * Cc + c) * Hh + h0 + tx];
  __syncthreads();
  const float g = gamma[0];
  float* dst = out + ((size_t)b * Cc + c) * HW;
  for (int i = ty; i < 32; i += 8)
    dst[(size_t)(h0 + i) * Ww + w0 + tx] += g * tile[tx][i];
}

// ---------------------------------------------------------------------------
extern "C" void kernel_launch(void* const* d_in, const int* in_sizes, int n_in,
                              void* d_out, int out_size, void* d_ws, size_t ws_size,
                              hipStream_t stream) {
  const float* x     = (const float*)d_in[0];
  const float* Wq    = (const float*)d_in[1];
  const float* bq    = (const float*)d_in[2];
  const float* Wk    = (const float*)d_in[3];
  const float* bk    = (const float*)d_in[4];
  const float* Wv    = (const float*)d_in[5];
  const float* bv    = (const float*)d_in[6];
  const float* gamma = (const float*)d_in[7];
  float* out = (float*)d_out;

  char* ws = (char*)d_ws;
  size_t off = 0;
  _Float16* qt   = (_Float16*)(ws + off); off += (size_t)Bn * HW * CQK * 2;        // 18.9 MB
  _Float16* kt   = (_Float16*)(ws + off); off += (size_t)Bn * HW * CQK * 2;        // 18.9 MB
  _Float16* vstd = (_Float16*)(ws + off); off += (size_t)Bn * Cc * HW * 2;         // 151 MB
  _Float16* vt   = (_Float16*)(ws + off); off += (size_t)Bn * Ww * Cc * Hh * 2;    // 151 MB
  float*    eHt  = (float*)(ws + off);    off += (size_t)Bn * Ww * Hh * 96 * 4;    // 56.6 MB
  float*    eWt  = (float*)(ws + off);    off += (size_t)Bn * Hh * Ww * 96 * 4;    // 56.6 MB
  _Float16* aHt  = (_Float16*)(ws + off); off += (size_t)Bn * Ww * Hh * 96 * 2;    // 28.3 MB
  _Float16* aWt  = (_Float16*)(ws + off); off += (size_t)Bn * Hh * Ww * 96 * 2;    // 28.3 MB
  float*    tmpH = (float*)(ws + off);    off += (size_t)Bn * Ww * Cc * Hh * 4;    // 302 MB

  // 1) projections (WMMA f16 GEMMs); q+k share x stream, v has 8 M-blocks
  k_proj_qk<<<dim3(HW / 128, 2, Bn), 256, 0, stream>>>(x, Wq, bq, qt, Wk, bk, kt);
  k_proj_v <<<dim3(HW / 128, 8, Bn), 256, 0, stream>>>(x, Wv, bv, vstd, vt);
  // 2) criss-cross energies (WMMA)
  k_energy<<<dim3(96, Bn, 2), 128, 0, stream>>>(kt, qt, eHt, eWt);
  // 3) softmax over 192 logits / pixel (diagonal masked)
  k_softmax<<<dim3((Bn * HW) / 256), 256, 0, stream>>>(eHt, eWt, aHt, aWt);
  // 4) column aggregation -> tmpH (WMMA)
  k_aggH<<<dim3(Ww, Bn), 256, 0, stream>>>(vt, aHt, tmpH);
  // 5) row aggregation + residual -> out (WMMA)
  k_aggW<<<dim3(Hh, Bn), 256, 0, stream>>>(vstd, aWt, x, gamma, out);
  // 6) out += gamma * transpose(tmpH)
  k_transpose_add<<<dim3(3, 3, Bn * Cc), 256, 0, stream>>>(tmpH, gamma, out);
}
